// GenericMoeLayer_37555194036224
// MI455X (gfx1250) — matrix operand order repaired and checked by reading, not compiled, not run on hardware
//
#include <hip/hip_runtime.h>
#include <hip/hip_bf16.h>
#include <math.h>

// Problem constants (from reference)
#define TOKENS   512
#define HIDDEN   1024
#define FFN      1408
#define NEXPERTS 8
// TOP_K = 2 (hardcoded in router logic)

typedef __attribute__((ext_vector_type(16))) __bf16 v16bf;
typedef __attribute__((ext_vector_type(8)))  float  v8f;

__device__ __forceinline__ float fast_silu(float g) {
    // g * sigmoid(g), with v_rcp_f32 instead of a correctly-rounded divide
    return g * __builtin_amdgcn_rcpf(1.f + __expf(-g));
}

// ---------------------------------------------------------------------------
// Kernel 0: zero output accumulator + per-expert counters
// ---------------------------------------------------------------------------
__global__ void moe_zero(float* __restrict__ out, int* __restrict__ cnt) {
    int i = blockIdx.x * blockDim.x + threadIdx.x;
    float4 z = {0.f, 0.f, 0.f, 0.f};
    if (i < (TOKENS * HIDDEN) / 4) ((float4*)out)[i] = z;
    if (i < NEXPERTS) cnt[i] = 0;
}

// ---------------------------------------------------------------------------
// Kernel 1: router. One wave (32 lanes) per token.
// logits -> top-2 of softmax -> renormalize (softmax denom cancels) ->
// append token into the two chosen experts' lists with its combine weight.
// ---------------------------------------------------------------------------
__global__ void moe_router(const float* __restrict__ x,
                           const float* __restrict__ gw,
                           int* __restrict__ cnt,
                           int* __restrict__ list,
                           float* __restrict__ wlist) {
    const int wave = (blockIdx.x * blockDim.x + threadIdx.x) >> 5;
    const int lane = threadIdx.x & 31;
    if (wave >= TOKENS) return;

    const float* xr = x + (size_t)wave * HIDDEN;
    float acc[NEXPERTS];
#pragma unroll
    for (int e = 0; e < NEXPERTS; ++e) acc[e] = 0.f;

    for (int h = lane; h < HIDDEN; h += 32) {
        float xv = xr[h];
#pragma unroll
        for (int e = 0; e < NEXPERTS; ++e)
            acc[e] = fmaf(xv, gw[e * HIDDEN + h], acc[e]);
    }
    // wave32 reduction
#pragma unroll
    for (int e = 0; e < NEXPERTS; ++e) {
#pragma unroll
        for (int off = 16; off > 0; off >>= 1)
            acc[e] += __shfl_xor(acc[e], off, 32);
    }

    if (lane == 0) {
        float mx = acc[0];
#pragma unroll
        for (int e = 1; e < NEXPERTS; ++e) mx = fmaxf(mx, acc[e]);
        float p[NEXPERTS];
#pragma unroll
        for (int e = 0; e < NEXPERTS; ++e) p[e] = __expf(acc[e] - mx);
        // top-2 (softmax denominator cancels under renormalization)
        int e0 = 0; float b0 = p[0];
#pragma unroll
        for (int e = 1; e < NEXPERTS; ++e) if (p[e] > b0) { b0 = p[e]; e0 = e; }
        int e1 = -1; float b1 = -1.f;
#pragma unroll
        for (int e = 0; e < NEXPERTS; ++e)
            if (e != e0 && p[e] > b1) { b1 = p[e]; e1 = e; }
        float inv = __builtin_amdgcn_rcpf(b0 + b1);
        int s0 = atomicAdd(&cnt[e0], 1);
        list[e0 * TOKENS + s0]  = wave;
        wlist[e0 * TOKENS + s0] = b0 * inv;
        int s1 = atomicAdd(&cnt[e1], 1);
        list[e1 * TOKENS + s1]  = wave;
        wlist[e1 * TOKENS + s1] = b1 * inv;
    }
}

// ---------------------------------------------------------------------------
// WMMA 16x16x32 bf16 operand layouts (ISA 7.12.2), half = (lane>=16):
//  A: lane l holds row m=l%16; elems j<8 -> K = half*8 + j,
//     elems j>=8 -> K = half*8 + 8 + j
//  B: lane l holds col n=l%16; elem j -> same K as A (table order)
//  D: VGPR r, lane l -> M = r + 8*half, N = l%16
// ---------------------------------------------------------------------------

// Kernel 2: per-expert gate/up GEMM (bf16 WMMA, fp32 acc) + SwiGLU,
// routing weight folded in; result stored as bf16 activations.
// Grid: (expert, FFN/16). Block: 128 threads = 4 waves.
// Each wave processes TWO 16-token M-tiles per iteration so each loaded
// B gate/up slice feeds 4 WMMAs (doubles WMMA density per byte loaded).
__global__ void moe_gemm1(const float* __restrict__ x,
                          const float* __restrict__ w1,
                          const int*   __restrict__ cnt,
                          const int*   __restrict__ list,
                          const float* __restrict__ wlist,
                          __bf16* __restrict__ act) {
    const int e     = blockIdx.x;
    const int nbase = blockIdx.y * 16;           // FFN tile base
    const int wave  = threadIdx.x >> 5;
    const int lane  = threadIdx.x & 31;
    const int half  = lane >> 4;
    const int l16   = lane & 15;

    const int count = cnt[e];
    const float* w1e = w1 + (size_t)e * (2 * FFN) * HIDDEN;
    const float* bg  = w1e + (size_t)(nbase + l16) * HIDDEN + half * 8;        // gate row
    const float* bu  = w1e + (size_t)(FFN + nbase + l16) * HIDDEN + half * 8;  // up row

    for (int mp = wave; mp * 32 < count; mp += 4) {
        const int mbase = mp * 32;
        int m0 = mbase + l16;
        int m1 = mbase + 16 + l16;
        int t0 = list[e * TOKENS + ((m0 < count) ? m0 : (count - 1))];
        int t1 = list[e * TOKENS + ((m1 < count) ? m1 : (count - 1))];
        const float* arow0 = x + (size_t)t0 * HIDDEN + half * 8;
        const float* arow1 = x + (size_t)t1 * HIDDEN + half * 8;

        v8f accg0 = {}, accu0 = {}, accg1 = {}, accu1 = {};
        for (int kb = 0; kb < HIDDEN; kb += 32) {
            v16bf a0, a1;
            const float* ap0 = arow0 + kb;
            const float* ap1 = arow1 + kb;
#pragma unroll
            for (int j = 0; j < 8; ++j) { a0[j] = (__bf16)ap0[j];      a1[j] = (__bf16)ap1[j]; }
#pragma unroll
            for (int j = 0; j < 8; ++j) { a0[8+j] = (__bf16)ap0[16+j]; a1[8+j] = (__bf16)ap1[16+j]; }

            v16bf bgv, buv;
            const float* bp = bg + kb;
            const float* bq = bu + kb;
#pragma unroll
            for (int j = 0; j < 16; ++j) bgv[j] = (__bf16)bp[j < 8 ? j : j + 8];
#pragma unroll
            for (int j = 0; j < 16; ++j) buv[j] = (__bf16)bq[j < 8 ? j : j + 8];

            accg0 = __builtin_amdgcn_wmma_f32_16x16x32_bf16(
                false, a0, false, bgv, (short)0, accg0, false, false);
            accu0 = __builtin_amdgcn_wmma_f32_16x16x32_bf16(
                false, a0, false, buv, (short)0, accu0, false, false);
            accg1 = __builtin_amdgcn_wmma_f32_16x16x32_bf16(
                false, a1, false, bgv, (short)0, accg1, false, false);
            accu1 = __builtin_amdgcn_wmma_f32_16x16x32_bf16(
                false, a1, false, buv, (short)0, accu1, false, false);
        }
#pragma unroll
        for (int r = 0; r < 8; ++r) {
            int row0 = mbase + r + half * 8;
            int row1 = row0 + 16;
            if (row0 < count) {
                float v = fast_silu(accg0[r]) * accu0[r] * wlist[e * TOKENS + row0];
                act[((size_t)e * TOKENS + row0) * FFN + nbase + l16] = (__bf16)v;
            }
            if (row1 < count) {
                float v = fast_silu(accg1[r]) * accu1[r] * wlist[e * TOKENS + row1];
                act[((size_t)e * TOKENS + row1) * FFN + nbase + l16] = (__bf16)v;
            }
        }
    }
}

// Kernel 3: per-expert down-projection (bf16 WMMA) + scatter-add into out.
// Grid: (expert, HIDDEN/16). Block: 128 threads = 4 waves.
// Two M-tiles per iteration: each loaded w2 slice feeds 2 WMMAs.
__global__ void moe_gemm2(const __bf16* __restrict__ act,
                          const float*  __restrict__ w2,
                          const int*    __restrict__ cnt,
                          const int*    __restrict__ list,
                          float* __restrict__ out) {
    const int e     = blockIdx.x;
    const int nbase = blockIdx.y * 16;           // HIDDEN tile base
    const int wave  = threadIdx.x >> 5;
    const int lane  = threadIdx.x & 31;
    const int half  = lane >> 4;
    const int l16   = lane & 15;

    const int count = cnt[e];
    const float* w2e  = w2 + (size_t)e * HIDDEN * FFN;
    const float* brow = w2e + (size_t)(nbase + l16) * FFN + half * 8;

    for (int mp = wave; mp * 32 < count; mp += 4) {
        const int mbase = mp * 32;
        const __bf16* arow0 =
            act + ((size_t)e * TOKENS + mbase + l16) * FFN + half * 8;
        const __bf16* arow1 = arow0 + (size_t)16 * FFN;

        v8f acc0 = {}, acc1 = {};
        for (int kb = 0; kb < FFN; kb += 32) {
            v16bf a0, a1;
            const __bf16* ap0 = arow0 + kb;
            const __bf16* ap1 = arow1 + kb;
#pragma unroll
            for (int j = 0; j < 8; ++j) { a0[j] = ap0[j];       a1[j] = ap1[j]; }
#pragma unroll
            for (int j = 0; j < 8; ++j) { a0[8+j] = ap0[16+j];  a1[8+j] = ap1[16+j]; }

            v16bf b;
            const float* bp = brow + kb;
#pragma unroll
            for (int j = 0; j < 16; ++j) b[j] = (__bf16)bp[j < 8 ? j : j + 8];

            acc0 = __builtin_amdgcn_wmma_f32_16x16x32_bf16(
                false, a0, false, b, (short)0, acc0, false, false);
            acc1 = __builtin_amdgcn_wmma_f32_16x16x32_bf16(
                false, a1, false, b, (short)0, acc1, false, false);
        }
#pragma unroll
        for (int r = 0; r < 8; ++r) {
            int row0 = mbase + r + half * 8;
            int row1 = row0 + 16;
            if (row0 < count) {
                int tok = list[e * TOKENS + row0];
                atomicAdd(&out[(size_t)tok * HIDDEN + nbase + l16], acc0[r]);
            }
            if (row1 < count) {
                int tok = list[e * TOKENS + row1];
                atomicAdd(&out[(size_t)tok * HIDDEN + nbase + l16], acc1[r]);
            }
        }
    }
}

// ---------------------------------------------------------------------------
// Host-side launcher
// ---------------------------------------------------------------------------
extern "C" void kernel_launch(void* const* d_in, const int* in_sizes, int n_in,
                              void* d_out, int out_size, void* d_ws, size_t ws_size,
                              hipStream_t stream) {
    const float* x   = (const float*)d_in[0];   // [512,1024]
    const float* gw  = (const float*)d_in[1];   // [8,1024]
    const float* w1  = (const float*)d_in[2];   // [8,2816,1024]
    const float* w2  = (const float*)d_in[3];   // [8,1024,1408]
    float*       out = (float*)d_out;           // [512,1024]

    // Workspace layout (bytes):
    //   [0,   64)                 : int cnt[8]
    //   [64,  64+16384)           : int list[8*512]
    //   [16448, 16448+16384)      : float wlist[8*512]
    //   [32832, +8*512*1408*2)    : bf16 act  (~11.5 MB)
    char* ws = (char*)d_ws;
    int*    cnt   = (int*)ws;
    int*    list  = (int*)(ws + 64);
    float*  wlist = (float*)(ws + 64 + 16384);
    __bf16* act   = (__bf16*)(ws + 64 + 16384 + 16384);

    // 0) zero out + counters
    moe_zero<<<(TOKENS * HIDDEN / 4 + 255) / 256, 256, 0, stream>>>(out, cnt);

    // 1) router: one wave per token, 4 waves per block
    moe_router<<<TOKENS / 4, 128, 0, stream>>>(x, gw, cnt, list, wlist);

    // 2) gate/up GEMM + SwiGLU -> bf16 act
    moe_gemm1<<<dim3(NEXPERTS, FFN / 16), 128, 0, stream>>>(
        x, w1, cnt, list, wlist, act);

    // 3) down projection + weighted scatter-add
    moe_gemm2<<<dim3(NEXPERTS, HIDDEN / 16), 128, 0, stream>>>(
        act, w2, cnt, list, out);
}
